// MultiHeadSelfAttention_72627896975366
// MI455X (gfx1250) — matrix operand order repaired and checked by reading, not compiled, not run on hardware
//
#include <hip/hip_runtime.h>

// ---------------- problem constants (from reference) ----------------
#define NB     4
#define SEQ    2048
#define EMBED  1024
#define HEADS  16
#define HD     64          // head dim

typedef __attribute__((ext_vector_type(16))) _Float16 v16h;
typedef __attribute__((ext_vector_type(8)))  _Float16 v8h;
typedef __attribute__((ext_vector_type(8)))  float    v8f;
typedef __attribute__((ext_vector_type(8)))  float    v8fv;

typedef __attribute__((ext_vector_type(4))) unsigned int u32x4;
typedef __attribute__((ext_vector_type(4))) int          i32x4;
typedef __attribute__((ext_vector_type(8))) int          i32x8;

#if __has_builtin(__builtin_amdgcn_tensor_load_to_lds)
#define HAVE_TDM 1
#endif

// D = A(16x32 f16) * B(32x16 f16) + C(16x16 f32), wave32.
__device__ __forceinline__ v8f wmma_f16(v16h a, v16h b, v8f c) {
    return __builtin_amdgcn_wmma_f32_16x16x32_f16(
        /*neg_a=*/false, a, /*neg_b=*/false, b,
        /*c_mod=*/(short)0, c, /*reuse_a=*/false, /*reuse_b=*/false);
}

#ifdef HAVE_TDM
// Issue a TDM load of a 32x64-halfword 2D tile (row stride EMBED halves)
// from global `ga` into LDS byte offset `ldsOff`. Wave-level op (EXEC ignored).
__device__ __forceinline__ void tdm_load_tile(unsigned long long ga, unsigned ldsOff) {
    u32x4 g0;
    g0[0] = 1u;                                            // count=1 (valid user D#)
    g0[1] = ldsOff;                                        // lds_addr (bytes)
    g0[2] = (unsigned)(ga & 0xFFFFFFFFu);                  // global_addr[31:0]
    g0[3] = (unsigned)((ga >> 32) & 0x01FFFFFFu)           // global_addr[56:32]
          | (2u << 30);                                    // type = 2 ("image")
    i32x8 g1;
    g1[0] = 0x00010000;           // wg_mask=0, data_size=1 (2 bytes)
    g1[1] = (64 << 16);           // tensor_dim0[15:0] = 64  (bits 79:48)
    g1[2] = (32 << 16);           // tensor_dim1[15:0] = 32  (bits 111:80)
    g1[3] = (64 << 16);           // tile_dim0 = 64          (bits 127:112)
    g1[4] = 32;                   // tile_dim1 = 32, tile_dim2 = 0
    g1[5] = EMBED;                // tensor_dim0_stride = 1024 (bits 207:160)
    g1[6] = 0;
    g1[7] = 0;
    i32x4 z4 = {0, 0, 0, 0};
#if defined(__clang_major__) && __clang_major__ >= 23
    i32x8 z8 = {0, 0, 0, 0, 0, 0, 0, 0};
    __builtin_amdgcn_tensor_load_to_lds(g0, g1, z4, z4, z8, 0);
#else
    __builtin_amdgcn_tensor_load_to_lds(g0, g1, z4, z4, 0);
#endif
}
#endif

// =====================================================================
// Kernel 0: one-shot f32 -> f16 conversion (weights), 8 elems/thread.
// =====================================================================
__global__ __launch_bounds__(256) void f32to16(const float* __restrict__ x,
                                               _Float16* __restrict__ y) {
    const size_t i = ((size_t)blockIdx.x * 256 + threadIdx.x) * 8;
    v8fv v = *(const v8fv*)(x + i);
    v8h  o;
#pragma unroll
    for (int j = 0; j < 8; ++j) o[j] = (_Float16)v[j];
    *(v8h*)(y + i) = o;
}

// =====================================================================
// Kernel 1: per-head projection  Y(r, e) = sum_d X(r, d) * W16(e, d)
// X: (R=NB*SEQ*HEADS, 64) fp32 contiguous; W16: (64,64) f16; Y: (R,64) f16.
// =====================================================================
__global__ __launch_bounds__(128) void proj64(const float* __restrict__ X,
                                              const _Float16* __restrict__ W16,
                                              _Float16* __restrict__ Y) {
    const int lane = threadIdx.x & 31;
    const int wave = threadIdx.x >> 5;
    const int half = lane >> 4;
    const int l16  = lane & 15;
    const long rbase = (long)blockIdx.x * 64 + wave * 16;

    // A-layout (16-bit, 16x32): lane m=l16 holds row m;
    //   elem e -> K = half*16 + e (contiguous, matches B)
    v16h A[2];
    {
        const float* xr = X + (rbase + l16) * 64;
#pragma unroll
        for (int ks = 0; ks < 2; ++ks) {
            v8fv lo = *(const v8fv*)(xr + ks * 32 + half * 16);
            v8fv hi = *(const v8fv*)(xr + ks * 32 + half * 16 + 8);
#pragma unroll
            for (int i = 0; i < 8; ++i) {
                A[ks][i]     = (_Float16)lo[i];
                A[ks][i + 8] = (_Float16)hi[i];
            }
        }
    }

#pragma unroll
    for (int t = 0; t < 4; ++t) {          // output cols t*16 .. t*16+15
        v8f c = {};
#pragma unroll
        for (int ks = 0; ks < 2; ++ks) {
            // B[d][e_out] = W[e_out][d] -> W16 row (t*16+l16), contiguous.
            v16h b = *(const v16h*)(W16 + (t * 16 + l16) * 64 + ks * 32 + half * 16);
            c = wmma_f16(A[ks], b, c);
        }
        _Float16* yo = Y + (rbase + 8 * half) * 64 + t * 16 + l16;
#pragma unroll
        for (int j = 0; j < 8; ++j)
            yo[(long)j * 64] = (_Float16)c[j];
    }
}

// =====================================================================
// Kernel 2: flash attention per (n, h, 64-query block), double-buffered.
// scores scaled by 1/sqrt(EMBED) = 1/32 (faithful to reference).
//   - K tile: Tensor Data Mover into ping-pong LDS (issued at top of the
//     iteration, TENSORcnt-waited at the bottom -> overlaps WMMA compute).
//   - V tile: global->regs at top, transposed regs->LDS at bottom.
// =====================================================================
__global__ __launch_bounds__(128) void attn(const _Float16* __restrict__ Qp,
                                            const _Float16* __restrict__ Kp,
                                            const _Float16* __restrict__ Vp,
                                            _Float16* __restrict__ Op) {
    const int QB   = SEQ / 64;                       // 32 query blocks
    const int qblk = blockIdx.x % QB;
    const int h    = (blockIdx.x / QB) % HEADS;
    const int n    = blockIdx.x / (QB * HEADS);

    const int lane = threadIdx.x & 31;
    const int wave = threadIdx.x >> 5;
    const int half = lane >> 4;
    const int l16  = lane & 15;

    __shared__ __attribute__((aligned(32))) _Float16 sK[2][32 * 64];   // keys x d
    __shared__ __attribute__((aligned(32))) _Float16 sVT[2][64 * 48];  // d x keys (padded)
    __shared__ __attribute__((aligned(32))) _Float16 sP[4][16 * 32];   // per-wave P tile

    const int q0 = qblk * 64 + wave * 16;            // wave's 16 query rows
    const int NT = SEQ / 32;                         // 64 key tiles

    // staging thread mapping: 32 key rows x 4 chunks of 16 halves
    const int r  = threadIdx.x >> 2;
    const int cq = threadIdx.x & 3;
    const _Float16* vbase = Vp + ((long)n * SEQ + r) * EMBED + h * HD + cq * 16;
#ifndef HAVE_TDM
    const _Float16* kbase = Kp + ((long)n * SEQ + r) * EMBED + h * HD + cq * 16;
#endif

    // Load this wave's Q tile once (A-layout), f16 from workspace.
    v16h Aq[2];
    {
        const _Float16* qr = Qp + ((long)n * SEQ + q0 + l16) * EMBED + h * HD;
#pragma unroll
        for (int ks = 0; ks < 2; ++ks) {
            v8h lo = *(const v8h*)(qr + ks * 32 + half * 16);
            v8h hi = *(const v8h*)(qr + ks * 32 + half * 16 + 8);
#pragma unroll
            for (int i = 0; i < 8; ++i) { Aq[ks][i] = lo[i]; Aq[ks][i + 8] = hi[i]; }
        }
    }

    v8f O0 = {}, O1 = {}, O2 = {}, O3 = {};
    float mrow[8], lrow[8];
#pragma unroll
    for (int j = 0; j < 8; ++j) { mrow[j] = -INFINITY; lrow[j] = 0.f; }

    const float scale = 0.03125f;                    // 1/sqrt(1024)

#ifdef HAVE_TDM
    const unsigned ldsK0 = (unsigned)(uintptr_t)(void*)&sK[0][0];  // low 32b of flat = LDS off
    const unsigned ldsK1 = (unsigned)(uintptr_t)(void*)&sK[1][0];
#endif

    // ---------------- preload tile 0 into buffer 0 ----------------
    {
#ifdef HAVE_TDM
        if (wave == 0)
            tdm_load_tile((unsigned long long)(uintptr_t)
                              (Kp + ((long)n * SEQ) * EMBED + h * HD),
                          ldsK0);
#else
        *(v8h*)(&sK[0][r * 64 + cq * 16])     = *(const v8h*)(kbase);
        *(v8h*)(&sK[0][r * 64 + cq * 16 + 8]) = *(const v8h*)(kbase + 8);
#endif
        v8h v0 = *(const v8h*)(vbase);
        v8h v1 = *(const v8h*)(vbase + 8);
#pragma unroll
        for (int i = 0; i < 8; ++i) {                // d-major (transposed) store
            sVT[0][(cq * 16 + i)     * 48 + r] = v0[i];
            sVT[0][(cq * 16 + 8 + i) * 48 + r] = v1[i];
        }
#ifdef HAVE_TDM
        if (wave == 0) __builtin_amdgcn_s_wait_tensorcnt(0);
#endif
    }
    __syncthreads();

    for (int kt = 0; kt < NT; ++kt) {
        const int  b        = kt & 1;
        const bool has_next = (kt + 1) < NT;

        // ---- kick off next tile's transfers (overlap with compute) ----
        v8h nv0, nv1;
#ifndef HAVE_TDM
        v8h nk0, nk1;
#endif
        if (has_next) {
            const long koff = (long)(kt + 1) * 32 * EMBED;
#ifdef HAVE_TDM
            if (wave == 0)
                tdm_load_tile((unsigned long long)(uintptr_t)
                                  (Kp + ((long)n * SEQ + (long)(kt + 1) * 32) * EMBED + h * HD),
                              b ? ldsK0 : ldsK1);
#else
            nk0 = *(const v8h*)(kbase + koff);
            nk1 = *(const v8h*)(kbase + koff + 8);
#endif
            nv0 = *(const v8h*)(vbase + koff);
            nv1 = *(const v8h*)(vbase + koff + 8);
        }

        // ---- scores: S0 = Q*K^T (keys 0-15), S1 = keys 16-31 ----
        const _Float16* kb = &sK[b][0];
        v8f S0 = {}, S1 = {};
#pragma unroll
        for (int ks = 0; ks < 2; ++ks) {
            v16h b0 = *(const v16h*)(kb + l16 * 64        + ks * 32 + half * 16);
            v16h b1 = *(const v16h*)(kb + (16 + l16) * 64 + ks * 32 + half * 16);
            S0 = wmma_f16(Aq[ks], b0, S0);
            S1 = wmma_f16(Aq[ks], b1, S1);
        }

        // ---- online softmax (row j+8*half lives in 16 lanes of this half) ----
#pragma unroll
        for (int j = 0; j < 8; ++j) {
            S0[j] *= scale; S1[j] *= scale;
            float t = fmaxf(S0[j], S1[j]);
            t = fmaxf(t, __shfl_xor(t, 1, 32));
            t = fmaxf(t, __shfl_xor(t, 2, 32));
            t = fmaxf(t, __shfl_xor(t, 4, 32));
            t = fmaxf(t, __shfl_xor(t, 8, 32));
            const float mn = fmaxf(mrow[j], t);
            const float al = __expf(mrow[j] - mn);
            mrow[j] = mn;
            S0[j] = __expf(S0[j] - mn);
            S1[j] = __expf(S1[j] - mn);
            float rs = S0[j] + S1[j];
            rs += __shfl_xor(rs, 1, 32);
            rs += __shfl_xor(rs, 2, 32);
            rs += __shfl_xor(rs, 4, 32);
            rs += __shfl_xor(rs, 8, 32);
            lrow[j] = lrow[j] * al + rs;
            O0[j] *= al; O1[j] *= al; O2[j] *= al; O3[j] *= al;
        }

        // ---- P: C-layout -> A-layout via per-wave LDS scratch ----
        _Float16* p = sP[wave];
#pragma unroll
        for (int j = 0; j < 8; ++j) {
            p[(j + 8 * half) * 32 + l16]      = (_Float16)S0[j];
            p[(j + 8 * half) * 32 + 16 + l16] = (_Float16)S1[j];
        }
        asm volatile("s_wait_dscnt 0" ::: "memory");  // same-wave LDS RAW
        v16h Pa;
        {
            v8h lo = *(const v8h*)(p + l16 * 32 + half * 16);
            v8h hi = *(const v8h*)(p + l16 * 32 + half * 16 + 8);
#pragma unroll
            for (int i = 0; i < 8; ++i) { Pa[i] = lo[i]; Pa[i + 8] = hi[i]; }
        }

        // ---- O += P(16x32) * V(32x64), B read contiguous from sVT ----
        const _Float16* vt = &sVT[b][0];
        v16h vb;
        vb = *(const v16h*)(vt + (0 * 16 + l16) * 48 + half * 16); O0 = wmma_f16(Pa, vb, O0);
        vb = *(const v16h*)(vt + (1 * 16 + l16) * 48 + half * 16); O1 = wmma_f16(Pa, vb, O1);
        vb = *(const v16h*)(vt + (2 * 16 + l16) * 48 + half * 16); O2 = wmma_f16(Pa, vb, O2);
        vb = *(const v16h*)(vt + (3 * 16 + l16) * 48 + half * 16); O3 = wmma_f16(Pa, vb, O3);

        // ---- finish staging next tile, then one barrier ----
        if (has_next) {
            _Float16* vtn = &sVT[1 - b][0];
#ifndef HAVE_TDM
            _Float16* kn = &sK[1 - b][0];
            *(v8h*)(kn + r * 64 + cq * 16)     = nk0;
            *(v8h*)(kn + r * 64 + cq * 16 + 8) = nk1;
#endif
#pragma unroll
            for (int i = 0; i < 8; ++i) {
                vtn[(cq * 16 + i)     * 48 + r] = nv0[i];
                vtn[(cq * 16 + 8 + i) * 48 + r] = nv1[i];
            }
#ifdef HAVE_TDM
            if (wave == 0) __builtin_amdgcn_s_wait_tensorcnt(0);
#endif
        }
        __syncthreads();   // readers of buf b done; buf 1-b fully staged
    }

    // ---- normalize and store f16 attention output ----
    float inv[8];
#pragma unroll
    for (int j = 0; j < 8; ++j) inv[j] = 1.f / lrow[j];
    _Float16* ob = Op + ((long)n * SEQ + q0 + 8 * half) * EMBED + h * HD + l16;
#pragma unroll
    for (int j = 0; j < 8; ++j) {
        _Float16* yo = ob + (long)j * EMBED;
        yo[0]  = (_Float16)(O0[j] * inv[j]);
        yo[16] = (_Float16)(O1[j] * inv[j]);
        yo[32] = (_Float16)(O2[j] * inv[j]);
        yo[48] = (_Float16)(O3[j] * inv[j]);
    }
}

// =====================================================================
// Kernel 3: output projection  Y = Oa(8192x1024 f16) @ Wo16^T + bo (fp32)
// =====================================================================
__global__ __launch_bounds__(128) void outproj(const _Float16* __restrict__ Oa,
                                               const _Float16* __restrict__ Wo16,
                                               const float* __restrict__ bo,
                                               float* __restrict__ Y) {
    const int MB   = (NB * SEQ) / 64;               // 128
    const int mblk = blockIdx.x % MB;
    const int nblk = blockIdx.x / MB;

    const int lane = threadIdx.x & 31;
    const int wave = threadIdx.x >> 5;
    const int half = lane >> 4;
    const int l16  = lane & 15;

    const long r0 = (long)mblk * 64 + wave * 16;
    const int  c0 = nblk * 64;

    v8f acc[4] = {{}, {}, {}, {}};

    for (int ks = 0; ks < EMBED / 32; ++ks) {
        const _Float16* ar = Oa + (r0 + l16) * EMBED + ks * 32;
        v16h a;
        {
            v8h lo = *(const v8h*)(ar + half * 16);
            v8h hi = *(const v8h*)(ar + half * 16 + 8);
#pragma unroll
            for (int i = 0; i < 8; ++i) { a[i] = lo[i]; a[i + 8] = hi[i]; }
        }
#pragma unroll
        for (int t = 0; t < 4; ++t) {
            v16h b = *(const v16h*)(Wo16 + (long)(c0 + t * 16 + l16) * EMBED
                                         + ks * 32 + half * 16);
            acc[t] = wmma_f16(a, b, acc[t]);
        }
    }

#pragma unroll
    for (int t = 0; t < 4; ++t) {
        const float bt = bo[c0 + t * 16 + l16];
        float* yo = Y + (r0 + 8 * half) * EMBED + c0 + t * 16 + l16;
#pragma unroll
        for (int j = 0; j < 8; ++j)
            yo[(long)j * EMBED] = acc[t][j] + bt;
    }
}

// =====================================================================
extern "C" void kernel_launch(void* const* d_in, const int* in_sizes, int n_in,
                              void* d_out, int out_size, void* d_ws, size_t ws_size,
                              hipStream_t stream) {
    const float* values  = (const float*)d_in[0];
    const float* keys    = (const float*)d_in[1];
    const float* queries = (const float*)d_in[2];
    /* d_in[3] = mask, unused by the reference forward */
    const float* Wv = (const float*)d_in[4];
    const float* Wk = (const float*)d_in[5];
    const float* Wq = (const float*)d_in[6];
    const float* Wo = (const float*)d_in[7];
    const float* bo = (const float*)d_in[8];
    float* out = (float*)d_out;

    // f16 workspace: Qp | Kp | Vp | Oa | Wo16 | Wv16 | Wk16 | Wq16
    // (4 x 16 MB + 2 MB + 24 KB ~= 69 MB; resident in the 192 MB L2).
    const size_t NSE = (size_t)NB * SEQ * EMBED;
    _Float16* Qp   = (_Float16*)d_ws;
    _Float16* Kp   = Qp + NSE;
    _Float16* Vp   = Kp + NSE;
    _Float16* Oa   = Vp + NSE;
    _Float16* Wo16 = Oa + NSE;
    _Float16* Wv16 = Wo16 + (size_t)EMBED * EMBED;
    _Float16* Wk16 = Wv16 + HD * HD;
    _Float16* Wq16 = Wk16 + HD * HD;
    (void)in_sizes; (void)n_in; (void)out_size; (void)ws_size;

    const dim3 blk(128);

    // one-shot weight conversions (f32 -> f16)
    f32to16<<<(EMBED * EMBED) / 2048, 256, 0, stream>>>(Wo, Wo16);
    f32to16<<<(HD * HD) / 2048, 256, 0, stream>>>(Wv, Wv16);
    f32to16<<<(HD * HD) / 2048, 256, 0, stream>>>(Wk, Wk16);
    f32to16<<<(HD * HD) / 2048, 256, 0, stream>>>(Wq, Wq16);

    // per-head projections
    proj64<<<(NB * SEQ * HEADS) / 64, blk, 0, stream>>>(values,  Wv16, Vp);
    proj64<<<(NB * SEQ * HEADS) / 64, blk, 0, stream>>>(keys,    Wk16, Kp);
    proj64<<<(NB * SEQ * HEADS) / 64, blk, 0, stream>>>(queries, Wq16, Qp);

    // flash attention (double-buffered TDM/async staging)
    attn<<<NB * HEADS * (SEQ / 64), blk, 0, stream>>>(Qp, Kp, Vp, Oa);

    // output projection + bias
    outproj<<<((NB * SEQ) / 64) * (EMBED / 64), blk, 0, stream>>>(Oa, Wo16, bo, out);
}
